// MLPRadianceField_47141561041208
// MI455X (gfx1250) — matrix operand order repaired and checked by reading, not compiled, not run on hardware
//
#include <hip/hip_runtime.h>
#include <hip/hip_bf16.h>

// ---------------------------------------------------------------------------
// NeRF MLP radiance field, fused inference for MI455X (gfx1250, wave32, WMMA)
//
//   * Weights converted once per launch f32 -> f16 into d_ws, pre-swizzled
//     into WMMA B-fragment order (16x16x32 f16 tiles).
//   * Main kernel: 256 threads = 8 wave32 per block, each wave owns a
//     16-point tile. Activations f16 in LDS, f32 accumulation in WMMA.
//   * Per-layer weights staged into a shared 128KB LDS buffer with the
//     gfx1250 async global->LDS path (ASYNCcnt / s_wait_asynccnt), overlapped
//     with compute where possible (global-weight skip layer, sigma head,
//     de-copy).
//   * Inner loop interleaves two N-tiles (two independent WMMA accumulator
//     chains) to hide the WMMA->WMMA data-hazard window (ISA 7.12.1).
//   * LDS: 2 x [128][320] f16 activations (160KB) + 128KB weight buffer.
// ---------------------------------------------------------------------------

typedef _Float16     v16h __attribute__((ext_vector_type(16)));
typedef _Float16     v8h  __attribute__((ext_vector_type(8)));
typedef float        v8f  __attribute__((ext_vector_type(8)));
typedef unsigned int v4u  __attribute__((ext_vector_type(4)));
typedef int          v4i  __attribute__((ext_vector_type(4)));

#define STR   320            // LDS activation row stride (f16 elements)
#define ROWS  128            // points per block
#define NWAVES 8
#define WBUF_BYTES (256 * 256 * 2)  // 128KB staged-weight buffer

#if defined(__has_builtin)
#  if __has_builtin(__builtin_amdgcn_global_load_async_to_lds_b128) && \
      __has_builtin(__builtin_amdgcn_s_wait_asynccnt)
#    define HAVE_ASYNC_LDS 1
#  endif
#endif
#ifndef HAVE_ASYNC_LDS
#  define HAVE_ASYNC_LDS 0
#endif

#if HAVE_ASYNC_LDS
// Builtin signature (verified via hipcc diagnostic): AS(1)/AS(3) v4i*.
typedef __attribute__((address_space(1))) v4i gas_v4i;
typedef __attribute__((address_space(3))) v4i las_v4i;
#endif

// ---------------------------------------------------------------------------
// Weight conversion: f32 [fanIn][fanOut] row-major -> f16 fragment-major.
//   dst[((kTile*NT + nTile)*32 + lane)*16 + j]  holds element
//   (k = kTile*32 + (lane>>4)*16 + j,  n = nTile*16 + (lane&15))   (contiguous k order, same as the A loads)
// ---------------------------------------------------------------------------
__global__ void nerf_convert_weights(const float* __restrict__ src,
                                     _Float16* __restrict__ dst,
                                     int fanIn, int fanOut, int KPAD, int NPAD) {
  int i = blockIdx.x * blockDim.x + threadIdx.x;
  int total = KPAD * NPAD;
  if (i >= total) return;
  int frag   = i >> 9;
  int within = i & 511;
  int lane   = within >> 4;
  int j      = within & 15;
  int ntiles = NPAD >> 4;
  int kTile  = frag / ntiles;
  int nTile  = frag - kTile * ntiles;
  int k = kTile * 32 + ((lane >> 4) << 4) + j;
  int n = nTile * 16 + (lane & 15);
  float v = (k < fanIn && n < fanOut) ? src[(size_t)k * fanOut + n] : 0.0f;
  dst[i] = (_Float16)v;
}

// ---------------------------------------------------------------------------
// Async staging of one layer's swizzled weights into LDS (issue/join split so
// the DMA can be overlapped with unrelated compute between the two calls).
// ---------------------------------------------------------------------------
__device__ __forceinline__ void stage_issue(const _Float16* __restrict__ gsrc,
                                            _Float16* ldst, int bytes, int tid) {
  const int nvec = bytes >> 4;                // 16B chunks
  for (int i = tid; i < nvec; i += 256) {
#if HAVE_ASYNC_LDS
    __builtin_amdgcn_global_load_async_to_lds_b128(
        (gas_v4i*)gsrc + i,
        (las_v4i*)ldst + i, 0, 0);
#else
    v4u t = *(const v4u*)((const char*)gsrc + (size_t)i * 16);
    *(v4u*)((char*)ldst + (size_t)i * 16) = t;
#endif
  }
}

__device__ __forceinline__ void stage_join() {
#if HAVE_ASYNC_LDS
  __builtin_amdgcn_s_wait_asynccnt(0);
#endif
  __syncthreads();
}

// ---------------------------------------------------------------------------
// One dense layer: [16 x K] (LDS f16) @ [K x N] (f16 fragments) -> LDS.
// A fragments hoisted into registers once; B fragments streamed per tile.
// Two N-tiles processed per iteration: the two independent accumulator
// chains alternate WMMA issue, covering the WMMA RAW hazard window.
// ---------------------------------------------------------------------------
template <int KSTEPS, int NTILES>
__device__ __forceinline__ void mlp_layer(const _Float16* __restrict__ wfrag,
                                          const float* __restrict__ bias,
                                          const _Float16* actIn,   // LDS tile base
                                          _Float16* actOut,        // LDS tile base
                                          int lane, bool relu) {
  const int g  = lane >> 4;
  const int nl = lane & 15;

  v16h afrag[KSTEPS];
#pragma unroll
  for (int k = 0; k < KSTEPS; ++k) {
    const _Float16* arow = actIn + nl * STR + k * 32 + g * 16;
    v8h a0 = *(const v8h*)(arow);        // K = k*32 + g*16 + 0..7
    v8h a1 = *(const v8h*)(arow + 8);    // K = k*32 + g*16 + 8..15
    afrag[k] = __builtin_shufflevector(a0, a1, 0, 1, 2, 3, 4, 5, 6, 7,
                                       8, 9, 10, 11, 12, 13, 14, 15);
  }

  static_assert((NTILES & 1) == 0, "NTILES must be even for 2-way interleave");
  for (int nTile = 0; nTile < NTILES; nTile += 2) {
    float bv0 = bias[nTile * 16 + nl];
    float bv1 = bias[(nTile + 1) * 16 + nl];
    v8f acc0 = {bv0, bv0, bv0, bv0, bv0, bv0, bv0, bv0};
    v8f acc1 = {bv1, bv1, bv1, bv1, bv1, bv1, bv1, bv1};
#pragma unroll
    for (int k = 0; k < KSTEPS; ++k) {
      v16h b0 = *(const v16h*)(wfrag + (((size_t)k * NTILES + nTile) * 32 + lane) * 16);
      v16h b1 = *(const v16h*)(wfrag + (((size_t)k * NTILES + nTile + 1) * 32 + lane) * 16);
      acc0 = __builtin_amdgcn_wmma_f32_16x16x32_f16(
          false, afrag[k], false, b0, (short)0, acc0, false, false);
      acc1 = __builtin_amdgcn_wmma_f32_16x16x32_f16(
          false, afrag[k], false, b1, (short)0, acc1, false, false);
    }
#pragma unroll
    for (int r = 0; r < 8; ++r) {
      float v0 = acc0[r];
      float v1 = acc1[r];
      if (relu) { v0 = fmaxf(v0, 0.0f); v1 = fmaxf(v1, 0.0f); }
      actOut[(r + g * 8) * STR + nTile * 16 + nl]        = (_Float16)v0;
      actOut[(r + g * 8) * STR + (nTile + 1) * 16 + nl]  = (_Float16)v1;
    }
  }
}

struct NerfParams {
  const float* xyzs;
  const float* dirs;
  const float* bias[10];    // b1_b0..4, b2_b0..2, b3_b, c_b0
  const float* sigw;
  const float* sigb;
  const float* cw1;
  const float* cb1;
  const _Float16* w[10];    // swizzled f16 weights per layer (in d_ws)
  float* out;               // [3N color | N sigma]
  int n;
};

__launch_bounds__(256)
__global__ void nerf_mlp(NerfParams P) {
  extern __shared__ char dynsmem[];
  _Float16* bufA = (_Float16*)dynsmem;
  _Float16* bufB = bufA + (size_t)ROWS * STR;
  _Float16* wbuf = bufB + (size_t)ROWS * STR;   // 128KB staged weights

  const int tid     = threadIdx.x;
  const int lane    = tid & 31;
  const int wave    = tid >> 5;
  const int rowBase = wave * 16;
  const int pBase   = blockIdx.x * ROWS + rowBase;

  _Float16* A = bufA + (size_t)rowBase * STR;   // this wave's tile
  _Float16* B = bufB + (size_t)rowBase * STR;

  // ---- Stage L0 weights + encode (disjoint LDS regions, overlapped) ------
  stage_issue(P.w[0], wbuf, 64 * 256 * 2, tid);

  if (lane < 16) {
    const int p = pBase + lane;
    _Float16* ra = A + (size_t)lane * STR;
    _Float16* rb = B + (size_t)lane * STR;
    float x = 0.f, y = 0.f, z = 0.f, dx = 0.f, dy = 0.f, dz = 0.f;
    if (p < P.n) {
      x = P.xyzs[p * 3 + 0]; y = P.xyzs[p * 3 + 1]; z = P.xyzs[p * 3 + 2];
      dx = P.dirs[p * 3 + 0]; dy = P.dirs[p * 3 + 1]; dz = P.dirs[p * 3 + 2];
    }
    // xe (63 wide, pad 64) -> bufA cols 0..63, copy -> bufB cols 256..319
    ra[0] = (_Float16)x; ra[1] = (_Float16)y; ra[2] = (_Float16)z;
    rb[256] = ra[0]; rb[257] = ra[1]; rb[258] = ra[2];
    float f = 1.0f;
    for (int i = 0; i < 10; ++i) {
      int b0 = 3 + i * 6;
      ra[b0 + 0] = (_Float16)__sinf(f * x); ra[b0 + 1] = (_Float16)__sinf(f * y);
      ra[b0 + 2] = (_Float16)__sinf(f * z); ra[b0 + 3] = (_Float16)__cosf(f * x);
      ra[b0 + 4] = (_Float16)__cosf(f * y); ra[b0 + 5] = (_Float16)__cosf(f * z);
      rb[256 + b0 + 0] = ra[b0 + 0]; rb[256 + b0 + 1] = ra[b0 + 1];
      rb[256 + b0 + 2] = ra[b0 + 2]; rb[256 + b0 + 3] = ra[b0 + 3];
      rb[256 + b0 + 4] = ra[b0 + 4]; rb[256 + b0 + 5] = ra[b0 + 5];
      f *= 2.0f;
    }
    ra[63]  = (_Float16)0.0f;
    rb[319] = (_Float16)0.0f;
    // de (27 wide, pad 32) parked in bufA cols 256..287
    ra[256] = (_Float16)dx; ra[257] = (_Float16)dy; ra[258] = (_Float16)dz;
    f = 1.0f;
    for (int i = 0; i < 4; ++i) {
      int b0 = 256 + 3 + i * 6;
      ra[b0 + 0] = (_Float16)__sinf(f * dx); ra[b0 + 1] = (_Float16)__sinf(f * dy);
      ra[b0 + 2] = (_Float16)__sinf(f * dz); ra[b0 + 3] = (_Float16)__cosf(f * dx);
      ra[b0 + 4] = (_Float16)__cosf(f * dy); ra[b0 + 5] = (_Float16)__cosf(f * dz);
      f *= 2.0f;
    }
    for (int c = 283; c < 288; ++c) ra[c] = (_Float16)0.0f;
  }
  stage_join();

  // ---- Backbone 1: 64->256, 4x 256->256, relu (weights in LDS) -----------
  mlp_layer<2, 16>(wbuf, P.bias[0], A, B, lane, true);
  __syncthreads();
  stage_issue(P.w[1], wbuf, 256 * 256 * 2, tid); stage_join();
  mlp_layer<8, 16>(wbuf, P.bias[1], B, A, lane, true);
  __syncthreads();
  stage_issue(P.w[2], wbuf, 256 * 256 * 2, tid); stage_join();
  mlp_layer<8, 16>(wbuf, P.bias[2], A, B, lane, true);
  __syncthreads();
  stage_issue(P.w[3], wbuf, 256 * 256 * 2, tid); stage_join();
  mlp_layer<8, 16>(wbuf, P.bias[3], B, A, lane, true);
  __syncthreads();
  stage_issue(P.w[4], wbuf, 256 * 256 * 2, tid); stage_join();
  mlp_layer<8, 16>(wbuf, P.bias[4], A, B, lane, true);
  __syncthreads();

  // ---- Skip layer L5 (320->256) reads weights from global; overlap the
  //      async staging of L6's weights with its compute. --------------------
  stage_issue(P.w[6], wbuf, 256 * 256 * 2, tid);
  __builtin_prefetch(P.w[5] + (size_t)tid * 1024, 0, 1);
  mlp_layer<10, 16>(P.w[5], P.bias[5], B, A, lane, true);
  stage_join();
  mlp_layer<8, 16>(wbuf, P.bias[6], A, B, lane, true);
  __syncthreads();
  stage_issue(P.w[7], wbuf, 256 * 256 * 2, tid); stage_join();
  mlp_layer<8, 16>(wbuf, P.bias[7], B, A, lane, true);   // h in bufA
  __syncthreads();

  // ---- Sigma head (256->1) on VALU, overlapped with b3 weight staging ----
  stage_issue(P.w[8], wbuf, 256 * 256 * 2, tid);
  if (lane < 16) {
    const int p = pBase + lane;
    const _Float16* hr = A + (size_t)lane * STR;
    float s = 0.0f;
    for (int kk = 0; kk < 256; ++kk) s += (float)hr[kk] * P.sigw[kk];
    s += P.sigb[0];
    if (p < P.n) P.out[(size_t)3 * P.n + p] = s;
  }
  stage_join();

  // ---- b3: 256->256, no activation ---------------------------------------
  mlp_layer<8, 16>(wbuf, P.bias[8], A, B, lane, false);
  __syncthreads();

  // ---- Stage c0 weights; overlap with de-copy into feat concat -----------
  stage_issue(P.w[9], wbuf, 288 * 128 * 2, tid);
  {
    int r = lane & 15, cg = lane >> 4;
    const _Float16* s16 = A + (size_t)r * STR + 256 + cg * 16;
    _Float16* d16 = B + (size_t)r * STR + 256 + cg * 16;
    v8h t0 = *(const v8h*)s16;
    v8h t1 = *(const v8h*)(s16 + 8);
    *(v8h*)d16 = t0;
    *(v8h*)(d16 + 8) = t1;
  }
  stage_join();

  // ---- Color layer 0: 288->128, relu -------------------------------------
  mlp_layer<9, 8>(wbuf, P.bias[9], B, A, lane, true);

  // ---- Color layer 1: 128->3 + sigmoid, VALU ------------------------------
  if (lane < 16) {
    const int p = pBase + lane;
    const _Float16* cr = A + (size_t)lane * STR;
    for (int ch = 0; ch < 3; ++ch) {
      float s = 0.0f;
      for (int kk = 0; kk < 128; ++kk) s += (float)cr[kk] * P.cw1[kk * 3 + ch];
      s += P.cb1[ch];
      if (p < P.n) P.out[(size_t)p * 3 + ch] = 1.0f / (1.0f + __expf(-s));
    }
  }
}

// ---------------------------------------------------------------------------
extern "C" void kernel_launch(void* const* d_in, const int* in_sizes, int n_in,
                              void* d_out, int out_size, void* d_ws, size_t ws_size,
                              hipStream_t stream) {
  (void)n_in; (void)out_size; (void)ws_size;
  const int n = in_sizes[0] / 3;

  const int widx[10]  = { 2,   4,   6,   8,  10,  12,  14,  16,  20,  22 };
  const int bidx[10]  = { 3,   5,   7,   9,  11,  13,  15,  17,  21,  23 };
  const int fin [10]  = {63, 256, 256, 256, 256, 319, 256, 256, 256, 283 };
  const int fout[10]  = {256,256, 256, 256, 256, 256, 256, 256, 256, 128 };
  const int kpad[10]  = {64, 256, 256, 256, 256, 320, 256, 256, 256, 288 };
  const int npad[10]  = {256,256, 256, 256, 256, 256, 256, 256, 256, 128 };

  _Float16* ws = (_Float16*)d_ws;
  NerfParams P;
  P.xyzs = (const float*)d_in[0];
  P.dirs = (const float*)d_in[1];
  P.sigw = (const float*)d_in[18];
  P.sigb = (const float*)d_in[19];
  P.cw1  = (const float*)d_in[24];
  P.cb1  = (const float*)d_in[25];
  P.out  = (float*)d_out;
  P.n    = n;

  size_t off = 0;
  for (int l = 0; l < 10; ++l) {
    const float* src = (const float*)d_in[widx[l]];
    _Float16*    dst = ws + off;
    int total = kpad[l] * npad[l];
    nerf_convert_weights<<<(total + 255) / 256, 256, 0, stream>>>(
        src, dst, fin[l], fout[l], kpad[l], npad[l]);
    P.w[l]    = dst;
    P.bias[l] = (const float*)d_in[bidx[l]];
    off += (size_t)total;
  }

  const size_t ldsBytes = (size_t)2 * ROWS * STR * sizeof(_Float16) + WBUF_BYTES; // 288 KB
  (void)hipFuncSetAttribute((const void*)nerf_mlp,
                            hipFuncAttributeMaxDynamicSharedMemorySize,
                            (int)ldsBytes);
  const int blocks = (n + ROWS - 1) / ROWS;
  nerf_mlp<<<blocks, 32 * NWAVES, ldsBytes, stream>>>(P);
}
